// CVRPLowerModel_38946763440471
// MI455X (gfx1250) — matrix operand order repaired and checked by reading, not compiled, not run on hardware
//
#include <hip/hip_runtime.h>
#include <math.h>

// ---------------------------------------------------------------------------
// CDNA5 / gfx1250 implementation. wave32. All matrix ops via
// v_wmma_f32_16x16x32_bf16 (A/B bf16, C/D f32), software-pipelined loads.
// ---------------------------------------------------------------------------

typedef __attribute__((ext_vector_type(16))) __bf16 v16bf;
typedef __attribute__((ext_vector_type(8)))  float  v8f;

namespace {
constexpr int kB   = 128;
constexpr int kN   = 512;   // NNB + 2
constexpr int kNNB = 510;
constexpr int kD   = 128;
constexpr int kFF  = 512;
constexpr int kL   = 3;
constexpr float kSqrtD = 11.31370849898476f;
}

union FragBf {
  v16bf v;
  uint4 q[2];
  unsigned short u[16];
};

__device__ __forceinline__ unsigned short f2bf(float f) {
  union { float f; unsigned u; } x; x.f = f;
  unsigned r = x.u + 0x7FFFu + ((x.u >> 16) & 1u);   // round-to-nearest-even
  return (unsigned short)(r >> 16);
}

__device__ __forceinline__ v8f wmma_bf16(const FragBf& a, const FragBf& b, v8f c) {
  return __builtin_amdgcn_wmma_f32_16x16x32_bf16(false, a.v, false, b.v,
                                                 (short)0, c, false, false);
}

__device__ __forceinline__ void load_frag(FragBf& f, const unsigned short* p) {
  f.q[0] = *reinterpret_cast<const uint4*>(p);
  f.q[1] = *reinterpret_cast<const uint4*>(p + 8);
}

// ---------------------------------------------------------------------------
// Weight transpose + f32 -> bf16 convert:  dst[c*R + r] = src[r*C + c]
// ---------------------------------------------------------------------------
__global__ __launch_bounds__(256)
void transpose_cvt_kernel(const float* __restrict__ src,
                          unsigned short* __restrict__ dst,
                          int R, int C, int total) {
  int idx = blockIdx.x * blockDim.x + threadIdx.x;
  if (idx >= total) return;
  int r = idx / C, c = idx % C;
  dst[(size_t)c * R + r] = f2bf(src[(size_t)r * C + c]);
}

// ---------------------------------------------------------------------------
// Build x rows 0,1:  q_first/q_last = fln @ Wq_{first,last} + load*w_load
// ---------------------------------------------------------------------------
__global__ __launch_bounds__(128)
void setup_qrows_kernel(const float* __restrict__ xy,
                        const float* __restrict__ load,
                        const float* __restrict__ embW,
                        const float* __restrict__ embB,
                        const float* __restrict__ WqF,
                        const float* __restrict__ WqL,
                        const float* __restrict__ wload,
                        float* __restrict__ x,
                        unsigned short* __restrict__ xbf) {
  int b = blockIdx.x, which = blockIdx.y, d = threadIdx.x;
  __shared__ float fln[kD];
  float xy0 = xy[((size_t)b * kN + which) * 2 + 0];
  float xy1 = xy[((size_t)b * kN + which) * 2 + 1];
  fln[d] = xy0 * embW[d] + xy1 * embW[kD + d] + embB[d];
  __syncthreads();
  const float* W = which ? WqL : WqF;
  float acc = 0.f;
  #pragma unroll 8
  for (int i = 0; i < kD; ++i) acc += fln[i] * W[(size_t)i * kD + d];
  float ld = load[b]; if (ld == 0.f) ld = 1.f;
  float val = acc + ld * wload[d];
  size_t o = ((size_t)b * kN + which) * kD + d;
  x[o] = val; xbf[o] = f2bf(val);
}

// ---------------------------------------------------------------------------
// Build neighbor rows (n >= 2) of x
// ---------------------------------------------------------------------------
__global__ __launch_bounds__(256)
void setup_nb_kernel(const float* __restrict__ xy,
                     const float* __restrict__ load,
                     const float* __restrict__ dem,
                     const float* __restrict__ mask,
                     const float* __restrict__ embW,
                     const float* __restrict__ embB,
                     const float* __restrict__ wdem,
                     float* __restrict__ x,
                     unsigned short* __restrict__ xbf) {
  long idx = (long)blockIdx.x * blockDim.x + threadIdx.x;
  if (idx >= (long)kB * kNNB * kD) return;
  int d = idx % kD;
  int t = (idx / kD) % kNNB;
  int b = idx / ((long)kD * kNNB);
  int n = 2 + t;
  float val = 0.f;
  if (!(mask[(size_t)b * kN + n] < 0.f)) {
    float xy0 = xy[((size_t)b * kN + n) * 2 + 0];
    float xy1 = xy[((size_t)b * kN + n) * 2 + 1];
    float fln = xy0 * embW[d] + xy1 * embW[kD + d] + embB[d];
    float ld = load[b]; if (ld == 0.f) ld = 1.f;
    float dm = dem[(size_t)b * kNNB + t] / ld;
    dm = fminf(fmaxf(dm, 0.f), 1.f);
    val = fln + dm * wdem[d];
  }
  size_t o = ((size_t)b * kN + n) * kD + d;
  x[o] = val; xbf[o] = f2bf(val);
}

// ---------------------------------------------------------------------------
// Fused QKV GEMM: q/k/v[M x 128] = A[M x 128](bf16) * {Wq,Wk,Wv}T (bf16).
// A fragments for the whole K=128 stay resident and feed Q, K, V in
// SEQUENCE (only 4 accumulators live at a time -> no spills), so A-matrix
// traffic is cut 3x vs separate GEMMs while register pressure stays low.
// ---------------------------------------------------------------------------
__global__ __launch_bounds__(256)
void gemm_qkv_kernel(const unsigned short* __restrict__ A,
                     const unsigned short* __restrict__ Bq,
                     const unsigned short* __restrict__ Bk,
                     const unsigned short* __restrict__ Bv,
                     float* __restrict__ outQ,
                     float* __restrict__ outK,
                     float* __restrict__ outV) {
  int lane = threadIdx.x & 31;
  int wave = threadIdx.x >> 5;
  int m0 = (blockIdx.x * 8 + wave) * 16;
  int n0 = blockIdx.y * 64;

  int rowA = m0 + (lane & 15);
  int kbA = (lane & 16) ? 16 : 0;
  int kbB = (lane & 16) ? 16 : 0;
  const size_t aBase = (size_t)rowA * kD;

  // Preload all four A fragments (whole K=128 row slice: 256 bytes).
  FragBf fa[4];
  #pragma unroll
  for (int s = 0; s < 4; ++s) load_frag(fa[s], A + aBase + 32 * s + kbA);

  // Per-j column base offsets (K-step folds into the load offset field).
  size_t colOff[4];
  #pragma unroll
  for (int j = 0; j < 4; ++j)
    colOff[j] = (size_t)(n0 + j * 16 + (lane & 15)) * kD + kbB;

  const unsigned short* Bmats[3] = { Bq, Bk, Bv };
  float* outs[3] = { outQ, outK, outV };

  #pragma unroll
  for (int mat = 0; mat < 3; ++mat) {
    const unsigned short* Bm = Bmats[mat];
    v8f c[4] = {};
    #pragma unroll
    for (int s = 0; s < 4; ++s) {
      #pragma unroll
      for (int j = 0; j < 4; ++j) {
        FragBf fb;
        load_frag(fb, Bm + colOff[j] + 32 * s);
        c[j] = wmma_bf16(fa[s], fb, c[j]);
      }
    }
    float* outP = outs[mat];
    #pragma unroll
    for (int j = 0; j < 4; ++j) {
      int col = n0 + j * 16 + (lane & 15);
      #pragma unroll
      for (int r = 0; r < 8; ++r) {
        int row = m0 + r + ((lane & 16) ? 8 : 0);
        outP[(size_t)row * kD + col] = c[j][r];
      }
    }
  }
}

// ---------------------------------------------------------------------------
// Generic WMMA GEMM (used for FFN): out[M x Nc] = A[M x K] * Bt[Nc x K].
// epi 1: +bias, relu, bf16 out; epi 2: +bias, f32 out.
// Double-buffered fragments, K step 32, manual unroll-by-2 for pipelining.
// ---------------------------------------------------------------------------
__global__ __launch_bounds__(256)
void gemm_bf16_kernel(const unsigned short* __restrict__ A,
                      const unsigned short* __restrict__ Bt,
                      const float* __restrict__ bias,
                      float* __restrict__ outF,
                      unsigned short* __restrict__ outH,
                      int M, int K, int Nc, int epi) {
  int lane = threadIdx.x & 31;
  int wave = threadIdx.x >> 5;
  int m0 = (blockIdx.x * 8 + wave) * 16;
  int n0 = blockIdx.y * 64;
  if (m0 >= M) return;

  v8f c[4] = {};
  int rowA = m0 + (lane & 15);
  int kbA = (lane & 16) ? 16 : 0;
  int kbB = (lane & 16) ? 16 : 0;
  const size_t aBase = (size_t)rowA * K;

  size_t colOff[4];
  #pragma unroll
  for (int j = 0; j < 4; ++j)
    colOff[j] = (size_t)(n0 + j * 16 + (lane & 15)) * K + kbB;

  FragBf fa[2];
  FragBf fb[2][4];

  auto loadSet = [&](int kk, int buf) {
    load_frag(fa[buf], A + aBase + kk + kbA);
    #pragma unroll
    for (int j = 0; j < 4; ++j)
      load_frag(fb[buf][j], Bt + colOff[j] + kk);
  };

  loadSet(0, 0);
  // K is a multiple of 64 (128 or 512)
  for (int kk = 0; kk < K; kk += 64) {
    loadSet(kk + 32, 1);
    #pragma unroll
    for (int j = 0; j < 4; ++j) c[j] = wmma_bf16(fa[0], fb[0][j], c[j]);
    if (kk + 64 < K) loadSet(kk + 64, 0);
    #pragma unroll
    for (int j = 0; j < 4; ++j) c[j] = wmma_bf16(fa[1], fb[1][j], c[j]);
  }

  #pragma unroll
  for (int j = 0; j < 4; ++j) {
    int col = n0 + j * 16 + (lane & 15);
    float bv = (epi != 0) ? bias[col] : 0.f;
    #pragma unroll
    for (int r = 0; r < 8; ++r) {
      int row = m0 + r + ((lane & 16) ? 8 : 0);
      float val = c[j][r] + bv;
      if (epi == 1) {
        outH[(size_t)row * Nc + col] = f2bf(fmaxf(val, 0.f));
      } else {
        outF[(size_t)row * Nc + col] = val;
      }
    }
  }
}

// ---------------------------------------------------------------------------
// Column max of k over s axis: kmax[b,d] = max_s k[b,s,d]
// ---------------------------------------------------------------------------
__global__ __launch_bounds__(256)
void kmax_kernel(const float* __restrict__ kf, float* __restrict__ kmax) {
  int idx = blockIdx.x * blockDim.x + threadIdx.x;
  int b = idx / kD, d = idx % kD;
  float m = -3.0e38f;
  for (int s = 0; s < kN; ++s)
    m = fmaxf(m, kf[((size_t)b * kN + s) * kD + d]);
  kmax[idx] = m;
}

// ---------------------------------------------------------------------------
// Row max of attention logits: wmax[b,t] = max_s (scale*dist + mask)
// one wave per row
// ---------------------------------------------------------------------------
__global__ __launch_bounds__(256)
void wmax_kernel(const float* __restrict__ dist,
                 const float* __restrict__ mask,
                 const float* __restrict__ alphaAttn,
                 const int* __restrict__ nnum,
                 int layer, float* __restrict__ wmax) {
  int lane = threadIdx.x & 31;
  int row = blockIdx.x * 8 + (threadIdx.x >> 5);
  int b = row / kN, t = row % kN;
  float scale = -alphaAttn[layer] * log2f((float)nnum[b]);
  const float* dr = dist + ((size_t)b * kN + t) * kN;
  const float* mr = mask + (size_t)b * kN;
  float m = -3.0e38f;
  for (int s = lane; s < kN; s += 32)
    m = fmaxf(m, scale * dr[s] + mr[s]);
  #pragma unroll
  for (int off = 16; off > 0; off >>= 1)
    m = fmaxf(m, __shfl_xor(m, off, 32));
  if (lane == 0) wmax[row] = m;
}

// ---------------------------------------------------------------------------
// ek/ekv: e = exp(k - kmax[b,d]); write ek, e*v transposed to (b, d, s) bf16
// LDS tile transpose, 32x32 tiles
// ---------------------------------------------------------------------------
__global__ __launch_bounds__(256)
void ekv_kernel(const float* __restrict__ kf,
                const float* __restrict__ vf,
                const float* __restrict__ kmax,
                unsigned short* __restrict__ ekt,
                unsigned short* __restrict__ ekvt) {
  __shared__ float le[32][33];
  __shared__ float lv[32][33];
  int b = blockIdx.x, s0 = blockIdx.y * 32, d0 = blockIdx.z * 32;
  int dx = threadIdx.x & 31;
  int sy0 = (threadIdx.x >> 5) * 4;
  float km = kmax[(size_t)b * kD + d0 + dx];
  #pragma unroll
  for (int i = 0; i < 4; ++i) {
    int s = s0 + sy0 + i;
    size_t o = ((size_t)b * kN + s) * kD + d0 + dx;
    float e = __expf(kf[o] - km);
    le[sy0 + i][dx] = e;
    lv[sy0 + i][dx] = e * vf[o];
  }
  __syncthreads();
  int sx = threadIdx.x & 31;
  int dy0 = (threadIdx.x >> 5) * 4;
  #pragma unroll
  for (int i = 0; i < 4; ++i) {
    int d = d0 + dy0 + i;
    size_t o = (size_t)b * kD * kN + (size_t)d * kN + s0 + sx;
    ekt[o]  = f2bf(le[sx][dy0 + i]);
    ekvt[o] = f2bf(lv[sx][dy0 + i]);
  }
}

// ---------------------------------------------------------------------------
// Fused AAFM: per wave a 16(t) x 128(d) tile.
//   num = ew @ ekv, den = ew @ ek, out = sigmoid(q) * num / den
// ew tile built on the fly (never materialized); B frags double-buffered.
// ---------------------------------------------------------------------------
__global__ __launch_bounds__(128)
void attn_kernel(const unsigned short* __restrict__ ekt,
                 const unsigned short* __restrict__ ekvt,
                 const float* __restrict__ qf,
                 const float* __restrict__ dist,
                 const float* __restrict__ mask,
                 const float* __restrict__ wmax,
                 const float* __restrict__ alphaAttn,
                 const int* __restrict__ nnum,
                 int layer, float* __restrict__ aafm) {
  int lane = threadIdx.x & 31;
  int wave = threadIdx.x >> 5;
  int b = blockIdx.x >> 3;                 // kN/64 == 8 tiles per batch
  int t0 = (blockIdx.x & 7) * 64 + wave * 16;
  float scale = -alphaAttn[layer] * log2f((float)nnum[b]);

  int tA = t0 + (lane & 15);
  int kbA = (lane & 16) ? 16 : 0;
  int kbB = (lane & 16) ? 16 : 0;
  const float* drow = dist + ((size_t)b * kN + tA) * kN;
  const float* mrow = mask + (size_t)b * kN;
  float wm = wmax[(size_t)b * kN + tA];
  const unsigned short* ektB = ekt  + (size_t)b * kD * kN;
  const unsigned short* ekvB = ekvt + (size_t)b * kD * kN;

  v8f cn[8] = {};
  v8f cd[8] = {};

  for (int kk = 0; kk < kN; kk += 32) {
    // Build ew A-fragment: exp(scale*dist + mask - rowmax), vectorized loads.
    FragBf fa;
    const float4* dlo = reinterpret_cast<const float4*>(drow + kk + kbA);
    const float4* dhi = reinterpret_cast<const float4*>(drow + kk + kbA + 8);
    const float4* mlo = reinterpret_cast<const float4*>(mrow + kk + kbA);
    const float4* mhi = reinterpret_cast<const float4*>(mrow + kk + kbA + 8);
    #pragma unroll
    for (int h = 0; h < 2; ++h) {
      float4 dv = dlo[h], mv = mlo[h];
      fa.u[4 * h + 0] = f2bf(__expf(scale * dv.x + mv.x - wm));
      fa.u[4 * h + 1] = f2bf(__expf(scale * dv.y + mv.y - wm));
      fa.u[4 * h + 2] = f2bf(__expf(scale * dv.z + mv.z - wm));
      fa.u[4 * h + 3] = f2bf(__expf(scale * dv.w + mv.w - wm));
      float4 dw = dhi[h], mw = mhi[h];
      fa.u[8 + 4 * h + 0] = f2bf(__expf(scale * dw.x + mw.x - wm));
      fa.u[8 + 4 * h + 1] = f2bf(__expf(scale * dw.y + mw.y - wm));
      fa.u[8 + 4 * h + 2] = f2bf(__expf(scale * dw.z + mw.z - wm));
      fa.u[8 + 4 * h + 3] = f2bf(__expf(scale * dw.w + mw.w - wm));
    }

    // Double-buffered B fragments over the 8 d-tiles.
    FragBf fbv[2], fbe[2];
    auto loadBV = [&](int j, int buf) {
      size_t o = (size_t)(j * 16 + (lane & 15)) * kN + kk + kbB;
      load_frag(fbv[buf], ekvB + o);
      load_frag(fbe[buf], ektB + o);
    };
    loadBV(0, 0);
    #pragma unroll
    for (int j = 0; j < 8; j += 2) {
      loadBV(j + 1, 1);
      cn[j] = wmma_bf16(fa, fbv[0], cn[j]);
      cd[j] = wmma_bf16(fa, fbe[0], cd[j]);
      if (j + 2 < 8) loadBV(j + 2, 0);
      cn[j + 1] = wmma_bf16(fa, fbv[1], cn[j + 1]);
      cd[j + 1] = wmma_bf16(fa, fbe[1], cd[j + 1]);
    }
  }

  #pragma unroll
  for (int j = 0; j < 8; ++j) {
    int d = j * 16 + (lane & 15);
    #pragma unroll
    for (int r = 0; r < 8; ++r) {
      int t = t0 + r + ((lane & 16) ? 8 : 0);
      size_t o = ((size_t)b * kN + t) * kD + d;
      float qv = qf[o];
      float sig = 1.f / (1.f + __expf(-qv));
      aafm[o] = sig * cn[j][r] / cd[j][r];
    }
  }
}

// ---------------------------------------------------------------------------
// LayerNorm over D of (in1 + in2); writes f32 + bf16 copies. 1 block per row.
// ---------------------------------------------------------------------------
__global__ __launch_bounds__(128)
void ln_kernel(const float* __restrict__ in1,
               const float* __restrict__ in2,
               const float* __restrict__ g,
               const float* __restrict__ bb,
               float* __restrict__ outF,
               unsigned short* __restrict__ outH) {
  __shared__ float r1[kD];
  __shared__ float r2[kD];
  int row = blockIdx.x, d = threadIdx.x;
  size_t o = (size_t)row * kD + d;
  float xv = in1[o] + in2[o];
  r1[d] = xv; r2[d] = xv * xv;
  __syncthreads();
  for (int off = kD / 2; off > 0; off >>= 1) {
    if (d < off) { r1[d] += r1[d + off]; r2[d] += r2[d + off]; }
    __syncthreads();
  }
  float m = r1[0] * (1.f / kD);
  float var = r2[0] * (1.f / kD) - m * m;
  float rs = rsqrtf(var + 1e-5f);
  float y = (xv - m) * rs * g[d] + bb[d];
  outF[o] = y;
  outH[o] = f2bf(y);
}

// ---------------------------------------------------------------------------
// Final scoring: score = 10*tanh(qo.x/sqrtD + bias) + mask -> softmax/argmax
// One block per batch.
// ---------------------------------------------------------------------------
__global__ __launch_bounds__(256)
void final_kernel(const float* __restrict__ x,
                  const float* __restrict__ dist,
                  const float* __restrict__ mask,
                  const int* __restrict__ nnum,
                  const int* __restrict__ nidx,
                  const float* __restrict__ alphaCom,
                  float* __restrict__ out) {
  __shared__ float qo[kD];
  __shared__ float sval[512];
  __shared__ float rmax[256];
  __shared__ int   ridx[256];
  __shared__ float rsum[256];
  int b = blockIdx.x, tid = threadIdx.x;
  if (tid < kD)
    qo[tid] = x[((size_t)b * kN + 0) * kD + tid] + x[((size_t)b * kN + 1) * kD + tid];
  __syncthreads();
  float logn = log2f((float)nnum[b]);
  float ac = alphaCom[0];
  for (int n = tid; n < 512; n += 256) {
    float sc = -3.0e38f;
    if (n < kNNB) {
      const float* xr = x + ((size_t)b * kN + 2 + n) * kD;
      float dot = 0.f;
      #pragma unroll 8
      for (int d = 0; d < kD; ++d) dot += qo[d] * xr[d];
      float bias = ac * (-logn * dist[((size_t)b * kN + 1) * kN + 2 + n]);
      sc = 10.f * tanhf(dot / kSqrtD + bias) + mask[(size_t)b * kN + 2 + n];
    }
    sval[n] = sc;
  }
  __syncthreads();
  float m = sval[tid]; int mi = tid;
  float m2 = sval[tid + 256];
  if (m2 > m) { m = m2; mi = tid + 256; }
  rmax[tid] = m; ridx[tid] = mi;
  __syncthreads();
  for (int off = 128; off > 0; off >>= 1) {
    if (tid < off) {
      bool take = (rmax[tid + off] > rmax[tid]) ||
                  (rmax[tid + off] == rmax[tid] && ridx[tid + off] < ridx[tid]);
      if (take) { rmax[tid] = rmax[tid + off]; ridx[tid] = ridx[tid + off]; }
    }
    __syncthreads();
  }
  float mx = rmax[0]; int best = ridx[0];
  rsum[tid] = __expf(sval[tid] - mx) + __expf(sval[tid + 256] - mx);
  __syncthreads();
  for (int off = 128; off > 0; off >>= 1) {
    if (tid < off) rsum[tid] += rsum[tid + off];
    __syncthreads();
  }
  if (tid == 0) {
    out[b]      = (float)nidx[(size_t)b * kNNB + best];
    out[kB + b] = __expf(sval[best] - mx) / rsum[0];
  }
}

// ---------------------------------------------------------------------------
// Host orchestration
// ---------------------------------------------------------------------------
extern "C" void kernel_launch(void* const* d_in, const int* in_sizes, int n_in,
                              void* d_out, int out_size, void* d_ws, size_t ws_size,
                              hipStream_t stream) {
  (void)in_sizes; (void)n_in; (void)out_size; (void)ws_size;
  const float* xy       = (const float*)d_in[0];
  const float* load     = (const float*)d_in[1];
  const float* dem      = (const float*)d_in[2];
  const float* mask     = (const float*)d_in[3];
  const float* dist     = (const float*)d_in[4];
  const int*   nnum     = (const int*)  d_in[5];
  const int*   nidx     = (const int*)  d_in[6];
  const float* embW     = (const float*)d_in[7];
  const float* embB     = (const float*)d_in[8];
  const float* wload    = (const float*)d_in[9];
  const float* wdem     = (const float*)d_in[10];
  const float* WqF      = (const float*)d_in[11];
  const float* WqL      = (const float*)d_in[12];
  const float* alphaCom = (const float*)d_in[13];
  const float* Wq       = (const float*)d_in[14];
  const float* Wk       = (const float*)d_in[15];
  const float* Wv       = (const float*)d_in[16];
  const float* alphaAtt = (const float*)d_in[17];
  const float* ln1g     = (const float*)d_in[18];
  const float* ln1b     = (const float*)d_in[19];
  const float* ln2g     = (const float*)d_in[20];
  const float* ln2b     = (const float*)d_in[21];
  const float* ffW1     = (const float*)d_in[22];
  const float* ffb1     = (const float*)d_in[23];
  const float* ffW2     = (const float*)d_in[24];
  const float* ffb2     = (const float*)d_in[25];

  char* base = (char*)d_ws;
  size_t off = 0;
  auto alloc = [&](size_t bytes) -> char* {
    char* p = base + off;
    off += (bytes + 255) & ~(size_t)255;
    return p;
  };

  const size_t MR = (size_t)kB * kN;                 // 65536 rows
  float*          x    = (float*)alloc(MR * kD * 4);
  unsigned short* xbf  = (unsigned short*)alloc(MR * kD * 2);
  float*          qf   = (float*)alloc(MR * kD * 4); // also reused as ffn2 out
  char*           kv   = alloc(2 * MR * kD * 4);     // k,v f32; reused as act bf16
  float*          kf   = (float*)kv;
  float*          vf   = (float*)(kv + MR * kD * 4);
  unsigned short* act  = (unsigned short*)kv;        // MR*kFF bf16 == same bytes
  unsigned short* ekt  = (unsigned short*)alloc(MR * kD * 2);
  unsigned short* ekvt = (unsigned short*)alloc(MR * kD * 2);
  float*          hbuf = (float*)alloc(MR * kD * 4); // aafm, then h
  float*          kmx  = (float*)alloc((size_t)kB * kD * 4);
  float*          wmx  = (float*)alloc((size_t)kB * kN * 4);
  unsigned short* wqT  = (unsigned short*)alloc((size_t)kL * kD * kD * 2);
  unsigned short* wkT  = (unsigned short*)alloc((size_t)kL * kD * kD * 2);
  unsigned short* wvT  = (unsigned short*)alloc((size_t)kL * kD * kD * 2);
  unsigned short* w1T  = (unsigned short*)alloc((size_t)kL * kFF * kD * 2);
  unsigned short* w2T  = (unsigned short*)alloc((size_t)kL * kD * kFF * 2);

  // --- one-time weight transpose/convert to bf16 -------------------------
  for (int l = 0; l < kL; ++l) {
    int tDD = kD * kD, tDF = kD * kFF;
    transpose_cvt_kernel<<<(tDD + 255) / 256, 256, 0, stream>>>(
        Wq + (size_t)l * tDD, wqT + (size_t)l * tDD, kD, kD, tDD);
    transpose_cvt_kernel<<<(tDD + 255) / 256, 256, 0, stream>>>(
        Wk + (size_t)l * tDD, wkT + (size_t)l * tDD, kD, kD, tDD);
    transpose_cvt_kernel<<<(tDD + 255) / 256, 256, 0, stream>>>(
        Wv + (size_t)l * tDD, wvT + (size_t)l * tDD, kD, kD, tDD);
    transpose_cvt_kernel<<<(tDF + 255) / 256, 256, 0, stream>>>(
        ffW1 + (size_t)l * tDF, w1T + (size_t)l * tDF, kD, kFF, tDF);
    transpose_cvt_kernel<<<(tDF + 255) / 256, 256, 0, stream>>>(
        ffW2 + (size_t)l * tDF, w2T + (size_t)l * tDF, kFF, kD, tDF);
  }

  // --- build x (f32 + bf16) ----------------------------------------------
  setup_qrows_kernel<<<dim3(kB, 2), kD, 0, stream>>>(
      xy, load, embW, embB, WqF, WqL, wload, x, xbf);
  long nbTot = (long)kB * kNNB * kD;
  setup_nb_kernel<<<(unsigned)((nbTot + 255) / 256), 256, 0, stream>>>(
      xy, load, dem, mask, embW, embB, wdem, x, xbf);

  // --- transformer layers -------------------------------------------------
  for (int l = 0; l < kL; ++l) {
    // fused QKV (A read once)
    dim3 gD((unsigned)(MR / 128), kD / 64);
    gemm_qkv_kernel<<<gD, 256, 0, stream>>>(
        xbf, wqT + (size_t)l * kD * kD, wkT + (size_t)l * kD * kD,
        wvT + (size_t)l * kD * kD, qf, kf, vf);

    kmax_kernel<<<(kB * kD) / 256, 256, 0, stream>>>(kf, kmx);
    wmax_kernel<<<(kB * kN) / 8, 256, 0, stream>>>(dist, mask, alphaAtt, nnum, l, wmx);
    ekv_kernel<<<dim3(kB, kN / 32, kD / 32), 256, 0, stream>>>(kf, vf, kmx, ekt, ekvt);

    attn_kernel<<<kB * (kN / 64), 128, 0, stream>>>(
        ekt, ekvt, qf, dist, mask, wmx, alphaAtt, nnum, l, hbuf);

    // h = LN(x + aafm), also produce h in bf16 (into xbf, x_bf16 is dead)
    ln_kernel<<<(unsigned)MR, kD, 0, stream>>>(
        x, hbuf, ln1g + (size_t)l * kD, ln1b + (size_t)l * kD, hbuf, xbf);

    // FFN1: act = relu(h @ W1 + b1)  (bf16 out, aliases dead k/v)
    dim3 gF1((unsigned)(MR / 128), kFF / 64);
    gemm_bf16_kernel<<<gF1, 256, 0, stream>>>(
        xbf, w1T + (size_t)l * kFF * kD, ffb1 + (size_t)l * kFF,
        nullptr, act, (int)MR, kD, kFF, 1);

    // FFN2: ffo = act @ W2 + b2  (f32 out, aliases dead q)
    dim3 gF2((unsigned)(MR / 128), kD / 64);
    gemm_bf16_kernel<<<gF2, 256, 0, stream>>>(
        act, w2T + (size_t)l * kD * kFF, ffb2 + (size_t)l * kD,
        qf, nullptr, (int)MR, kFF, kD, 2);

    // x = LN(h + ffo), also refresh x_bf16 for next layer
    ln_kernel<<<(unsigned)MR, kD, 0, stream>>>(
        hbuf, qf, ln2g + (size_t)l * kD, ln2b + (size_t)l * kD, x, xbf);
  }

  // --- scoring / softmax / argmax -----------------------------------------
  final_kernel<<<kB, 256, 0, stream>>>(x, dist, mask, nnum, nidx, alphaCom,
                                       (float*)d_out);
}
